// SelfAttentionLayer_62517543960960
// MI455X (gfx1250) — compile-verified, matrix operand order checked by reading, not run on hardware
//
#include <hip/hip_runtime.h>

#define HIDDEN   1024
#define HEADS    16
#define HEAD_DIM 64
#define SEQ      2048
#define BATCH    2

typedef _Float16 v16h __attribute__((ext_vector_type(16)));
typedef _Float16 v8h  __attribute__((ext_vector_type(8)));
typedef float    v8f  __attribute__((ext_vector_type(8)));

// Load a 16-half WMMA A/B fragment per the CDNA5 16-bit layout:
// lane holds K = {half8+0..7} in regs 0..3 and K = {half8+16..23} in regs 4..7.
// 'p' must already point at (row_base + half8).
__device__ __forceinline__ v16h frag_ld(const _Float16* p) {
    v8h lo = *(const v8h*)(p);
    v8h hi = *(const v8h*)(p + 16);
    v16h r;
#pragma unroll
    for (int i = 0; i < 8; ++i) { r[i] = lo[i]; r[8 + i] = hi[i]; }
    return r;
}

__device__ __forceinline__ v8f wmma_f16(v16h a, v16h b, v8f c) {
    return __builtin_amdgcn_wmma_f32_16x16x32_f16(false, a, false, b, (short)0, c,
                                                  false, false);
}

// Async global -> LDS copy, 16 bytes per lane. Tracked by ASYNCcnt.
__device__ __forceinline__ void async_b128(uint32_t lds_off, const _Float16* g) {
    unsigned long long ga = (unsigned long long)(uintptr_t)g;
    asm volatile("global_load_async_to_lds_b128 %0, %1, off"
                 :: "v"(lds_off), "v"(ga) : "memory");
}

// ---------------------------------------------------------------------------
// Kernel 1: Y = (X @ W + bias) * scale, cast to f16.
//   VTRANS==0 -> out[B,H,S,D] (Q, K)
//   VTRANS==1 -> out[B,H,D,S] (V, pre-transposed so attention can async-copy)
// Grid: (HIDDEN/64 = 16 n-tiles, B*S/64 = 64 m-tiles). Block: 128 threads.
// ---------------------------------------------------------------------------
template <int VTRANS>
__global__ __launch_bounds__(128) void qkv_gemm(
    const float* __restrict__ X,     // [B*S, HIDDEN] f32
    const float* __restrict__ W,     // [HIDDEN, HIDDEN] f32
    const float* __restrict__ bias,  // [HIDDEN] f32
    _Float16* __restrict__ out,
    float scale)
{
    __shared__ _Float16 Xt[64][32];  // [m][k]
    __shared__ _Float16 Wt[64][32];  // transposed: [n][k]

    const int tid   = threadIdx.x;
    const int lane  = tid & 31;
    const int wave  = tid >> 5;
    const int l15   = lane & 15;
    const int half8 = (lane >> 4) * 8;
    const int mBase = blockIdx.y * 64;
    const int nBase = blockIdx.x * 64;

    v8f acc[4] = {};

    for (int k0 = 0; k0 < HIDDEN; k0 += 32) {
        __syncthreads();
        // X tile: 64x32 f32 -> f16, row-major. 512 float4 loads / 128 threads.
#pragma unroll
        for (int i = 0; i < 4; ++i) {
            int e    = tid + i * 128;      // 0..511
            int row  = e >> 3;             // 0..63
            int col4 = (e & 7) * 4;        // 0..28
            const float4 f = *(const float4*)(X + (size_t)(mBase + row) * HIDDEN + k0 + col4);
            Xt[row][col4 + 0] = (_Float16)f.x;
            Xt[row][col4 + 1] = (_Float16)f.y;
            Xt[row][col4 + 2] = (_Float16)f.z;
            Xt[row][col4 + 3] = (_Float16)f.w;
        }
        // W tile transposed: W[k0+k][nBase+n] -> Wt[n][k].
#pragma unroll
        for (int i = 0; i < 4; ++i) {
            int e    = tid + i * 128;
            int krow = e >> 4;             // 0..31
            int n4   = (e & 15) * 4;       // 0..60
            const float4 f = *(const float4*)(W + (size_t)(k0 + krow) * HIDDEN + nBase + n4);
            Wt[n4 + 0][krow] = (_Float16)f.x;
            Wt[n4 + 1][krow] = (_Float16)f.y;
            Wt[n4 + 2][krow] = (_Float16)f.z;
            Wt[n4 + 3][krow] = (_Float16)f.w;
        }
        __syncthreads();

        v16h a = frag_ld(&Xt[wave * 16 + l15][half8]);
#pragma unroll
        for (int ns = 0; ns < 4; ++ns) {
            v16h b = frag_ld(&Wt[ns * 16 + l15][half8]);
            acc[ns] = wmma_f16(a, b, acc[ns]);
        }
    }

    // Epilogue: bias, scale, cast, scatter. h == blockIdx.x (64-wide n tiles).
    // A 64-row m-tile never straddles a batch boundary (SEQ % 64 == 0), so the
    // batch index and row base are uniform per block.
    const int h    = blockIdx.x;
    const int bIdx = mBase >> 11;                       // mBase / SEQ
    const int s0   = (mBase & (SEQ - 1)) + wave * 16 + half8;
#pragma unroll
    for (int ns = 0; ns < 4; ++ns) {
        const int   d  = ns * 16 + l15;
        const float bv = bias[nBase + d];
        if (VTRANS) {
            _Float16* op = out + (((size_t)bIdx * HEADS + h) * HEAD_DIM + d) * SEQ + s0;
#pragma unroll
            for (int r = 0; r < 8; ++r)
                op[r] = (_Float16)((acc[ns][r] + bv) * scale);
        } else {
            _Float16* op = out + (((size_t)bIdx * HEADS + h) * SEQ + s0) * HEAD_DIM + d;
#pragma unroll
            for (int r = 0; r < 8; ++r)
                op[(size_t)r * HEAD_DIM] = (_Float16)((acc[ns][r] + bv) * scale);
        }
    }
}

// ---------------------------------------------------------------------------
// Kernel 2: flash attention with async double-buffered K/V tiles.
// Grid: (SEQ/64 q-tiles, B*HEADS). Block: 128 (4 waves, 16 q rows each).
// ---------------------------------------------------------------------------
__global__ __launch_bounds__(128) void attn(
    const _Float16* __restrict__ Q,   // [B*H, S, D] (pre-scaled by 1/sqrt(D))
    const _Float16* __restrict__ K,   // [B*H, S, D]
    const _Float16* __restrict__ VT,  // [B*H, D, S] (transposed values)
    const int* __restrict__ mask,     // [B, S]
    float* __restrict__ out)          // [B, S, HIDDEN]
{
    __shared__ _Float16 Kt[2][64][64];   // key tiles, row-major [k][d]
    __shared__ _Float16 Vt[2][64][64];   // value tiles, transposed [d][k]
    __shared__ _Float16 Pt[4][16][64];   // per-wave probability tile [q][k]

    const int tid   = threadIdx.x;
    const int lane  = tid & 31;
    const int wave  = tid >> 5;
    const int l15   = lane & 15;
    const int half8 = (lane >> 4) * 8;
    const int bh    = blockIdx.y;        // 0..31
    const int b     = bh >> 4;
    const int h     = bh & 15;
    const int qBase = blockIdx.x * 64 + wave * 16;

    const _Float16* Qp  = Q  + (size_t)bh * SEQ * HEAD_DIM;
    const _Float16* Kp  = K  + (size_t)bh * SEQ * HEAD_DIM;
    const _Float16* VTp = VT + (size_t)bh * HEAD_DIM * SEQ;
    const int*      mp  = mask + b * SEQ;

    // Q fragments for d=0..31 and d=32..63 (held in registers for whole loop).
    const _Float16* qrow = Qp + (size_t)(qBase + l15) * HEAD_DIM + half8;
    const v16h qa0 = frag_ld(qrow);
    const v16h qa1 = frag_ld(qrow + 32);

    float mrun[8], lrun[8];
    v8f acc[4] = {};
#pragma unroll
    for (int r = 0; r < 8; ++r) { mrun[r] = -3.0e38f; lrun[r] = 0.0f; }

    // Issue one 64-key tile (K rows + VT rows) as 8 async b128 ops per thread.
    auto issue_tile = [&](int kt, int buf) {
#pragma unroll
        for (int i = 0; i < 4; ++i) {
            int e   = tid + i * 128;     // 0..511
            int row = e >> 3;            // 0..63
            int c   = (e & 7) * 8;       // 0..56 halfs
            async_b128((uint32_t)(uintptr_t)&Kt[buf][row][c],
                       Kp + (size_t)(kt + row) * HEAD_DIM + c);
            async_b128((uint32_t)(uintptr_t)&Vt[buf][row][c],
                       VTp + (size_t)row * SEQ + kt + c);
        }
    };

    issue_tile(0, 0);

    for (int t = 0; t < SEQ / 64; ++t) {
        const int kt   = t * 64;
        const int bufc = t & 1;

        __syncthreads();  // all waves done reading buf[(t+1)&1] (tile t-1)
        if (t + 1 < SEQ / 64) {
            issue_tile(kt + 64, (t + 1) & 1);
            // In-order completion: <=8 in flight means tile t fully landed.
            asm volatile("s_wait_asynccnt 0x8" ::: "memory");
        } else {
            asm volatile("s_wait_asynccnt 0x0" ::: "memory");
        }
        __syncthreads();  // every wave has drained its own ASYNCcnt

        // Scores: four 16x16 key subtiles, K-depth 64 = 2 WMMAs each.
        v8f s[4] = {};
#pragma unroll
        for (int ks = 0; ks < 4; ++ks) {
            const _Float16* kr = &Kt[bufc][ks * 16 + l15][half8];
            s[ks] = wmma_f16(qa0, frag_ld(kr),      s[ks]);
            s[ks] = wmma_f16(qa1, frag_ld(kr + 32), s[ks]);
        }

        // Mask: C-fragment column == lane&15 across all 8 accumulator regs.
#pragma unroll
        for (int ks = 0; ks < 4; ++ks) {
            const int mv = mp[kt + ks * 16 + l15];
#pragma unroll
            for (int r = 0; r < 8; ++r) s[ks][r] = mv ? s[ks][r] : -1.0e8f;
        }

        // Online softmax over 64 keys (reductions across 16-lane halves).
        float mnew[8], sc[8];
#pragma unroll
        for (int r = 0; r < 8; ++r) {
            float tm = fmaxf(fmaxf(s[0][r], s[1][r]), fmaxf(s[2][r], s[3][r]));
            tm = fmaxf(tm, __shfl_xor(tm, 1, 32));
            tm = fmaxf(tm, __shfl_xor(tm, 2, 32));
            tm = fmaxf(tm, __shfl_xor(tm, 4, 32));
            tm = fmaxf(tm, __shfl_xor(tm, 8, 32));
            mnew[r] = fmaxf(mrun[r], tm);
            sc[r]   = __expf(mrun[r] - mnew[r]);
            mrun[r] = mnew[r];
        }
#pragma unroll
        for (int r = 0; r < 8; ++r) {
#pragma unroll
            for (int ks = 0; ks < 4; ++ks) s[ks][r] = __expf(s[ks][r] - mnew[r]);
            float ssum = (s[0][r] + s[1][r]) + (s[2][r] + s[3][r]);
            ssum += __shfl_xor(ssum, 1, 32);
            ssum += __shfl_xor(ssum, 2, 32);
            ssum += __shfl_xor(ssum, 4, 32);
            ssum += __shfl_xor(ssum, 8, 32);
            lrun[r] = lrun[r] * sc[r] + ssum;
#pragma unroll
            for (int ds = 0; ds < 4; ++ds) acc[ds][r] *= sc[r];
        }

        // Re-stripe P from C layout to A layout via per-wave LDS slab.
#pragma unroll
        for (int r = 0; r < 8; ++r) {
#pragma unroll
            for (int ks = 0; ks < 4; ++ks)
                Pt[wave][half8 + r][ks * 16 + l15] = (_Float16)s[ks][r];
        }
        asm volatile("s_wait_dscnt 0" ::: "memory");  // same-wave cross-lane LDS RAW

        // P @ V: two K-steps of 32 over the 64-key tile.
#pragma unroll
        for (int j = 0; j < 2; ++j) {
            v16h pa = frag_ld(&Pt[wave][l15][j * 32 + half8]);
#pragma unroll
            for (int ds = 0; ds < 4; ++ds) {
                v16h vb = frag_ld(&Vt[bufc][ds * 16 + l15][j * 32 + half8]);
                acc[ds] = wmma_f16(pa, vb, acc[ds]);
            }
        }
    }

    // Final 1/rowsum and store f32 to [B, S, h*64 + d].
#pragma unroll
    for (int r = 0; r < 8; ++r) {
        float inv = 1.0f / lrun[r];
        int   s   = qBase + half8 + r;
        float* orow = out + ((size_t)b * SEQ + s) * HIDDEN + h * HEAD_DIM;
#pragma unroll
        for (int ds = 0; ds < 4; ++ds) {
            orow[ds * 16 + l15] = acc[ds][r] * inv;
        }
    }
}

// ---------------------------------------------------------------------------
extern "C" void kernel_launch(void* const* d_in, const int* in_sizes, int n_in,
                              void* d_out, int out_size, void* d_ws, size_t ws_size,
                              hipStream_t stream) {
    (void)in_sizes; (void)n_in; (void)out_size; (void)ws_size;
    const float* X    = (const float*)d_in[0];
    const int*   mask = (const int*)d_in[1];
    const float* Wq   = (const float*)d_in[2];
    const float* bq   = (const float*)d_in[3];
    const float* Wk   = (const float*)d_in[4];
    const float* bk   = (const float*)d_in[5];
    const float* Wv   = (const float*)d_in[6];
    const float* bv   = (const float*)d_in[7];

    _Float16* ws  = (_Float16*)d_ws;
    const size_t per = (size_t)BATCH * HEADS * SEQ * HEAD_DIM;  // 4 Mi halfs = 8 MB
    _Float16* Qh  = ws;
    _Float16* Kh  = ws + per;
    _Float16* VTh = ws + 2 * per;

    dim3 g1(HIDDEN / 64, (BATCH * SEQ) / 64);   // (16, 64)
    qkv_gemm<0><<<g1, 128, 0, stream>>>(X, Wq, bq, Qh,  0.125f);  // fold 1/sqrt(64)
    qkv_gemm<0><<<g1, 128, 0, stream>>>(X, Wk, bk, Kh,  1.0f);
    qkv_gemm<1><<<g1, 128, 0, stream>>>(X, Wv, bv, VTh, 1.0f);    // transposed V

    dim3 g2(SEQ / 64, BATCH * HEADS);           // (32, 32)
    attn<<<g2, 128, 0, stream>>>(Qh, Kh, VTh, mask, (float*)d_out);
}
